// FHEBlock_7679401525971
// MI455X (gfx1250) — matrix operand order repaired and checked by reading, compiled, not run on hardware
//
#include <hip/hip_runtime.h>

// ---------------------------------------------------------------------------
// CDNA5 (gfx1250) FHE transformer block: everything is C = A * B^T in bf16
// via v_wmma_f32_16x16x32_bf16, with fused epilogues.
// Round 3: GLOBAL_LOAD_ASYNC_TO_LDS_B64 with correctly-typed pointer args
// (param type from compiler diagnostic: v2i32 addrspace(1)* / addrspace(3)*).
// ---------------------------------------------------------------------------

typedef __attribute__((ext_vector_type(16))) __bf16 v16bf;
typedef __attribute__((ext_vector_type(8)))  float  v8f;
typedef int v2i_gcc __attribute__((vector_size(8)));   // matches builtin param

#define TM 128
#define TN 128
#define TK 32
#define LROW 40   // TK + 8 pad elements; keeps all b64 LDS accesses 8B-aligned

#if defined(__HIP_DEVICE_COMPILE__) && \
    __has_builtin(__builtin_amdgcn_global_load_async_to_lds_b64) && \
    __has_builtin(__builtin_amdgcn_s_wait_asynccnt)
#define USE_ASYNC_LDS 1
#else
#define USE_ASYNC_LDS 0
#endif

#define AS1 __attribute__((address_space(1)))
#define AS3 __attribute__((address_space(3)))

enum { EPI_BF16 = 0, EPI_POLY = 1, EPI_BF16T = 2, EPI_BLEND = 3, EPI_BLEND_LN = 4 };

__device__ __forceinline__ unsigned short f2bf(float f) {
  unsigned int u = __float_as_uint(f);
  u += 0x7FFFu + ((u >> 16) & 1u);            // round-to-nearest-even
  return (unsigned short)(u >> 16);
}

// ---------------- elementwise prep kernels ----------------
__global__ void cvt_bf16_kernel(const float* __restrict__ in,
                                unsigned short* __restrict__ out, int n4) {
  int i = blockIdx.x * blockDim.x + threadIdx.x;
  if (i >= n4) return;
  float4 f = ((const float4*)in)[i];
  uint2 o;
  o.x = (unsigned)f2bf(f.x) | ((unsigned)f2bf(f.y) << 16);
  o.y = (unsigned)f2bf(f.z) | ((unsigned)f2bf(f.w) << 16);
  ((uint2*)out)[i] = o;
}

// out = bf16(x * lnw[col] * 0.1)
__global__ void scale_ln_bf16_kernel(const float* __restrict__ x,
                                     const float* __restrict__ lnw,
                                     unsigned short* __restrict__ out,
                                     int n4, int cmask4) {
  int i = blockIdx.x * blockDim.x + threadIdx.x;
  if (i >= n4) return;
  float4 f = ((const float4*)x)[i];
  float4 w = ((const float4*)lnw)[i & cmask4];
  uint2 o;
  o.x = (unsigned)f2bf(f.x * w.x * 0.1f) | ((unsigned)f2bf(f.y * w.y * 0.1f) << 16);
  o.y = (unsigned)f2bf(f.z * w.z * 0.1f) | ((unsigned)f2bf(f.w * w.w * 0.1f) << 16);
  ((uint2*)out)[i] = o;
}

// ---------------- WMMA GEMM: C = epilogue( alpha * A[M,K] * B[N,K]^T ) ------
// 256 threads = 8 wave32 in 2x4 grid; wave tile 64x32 = 4x2 WMMA 16x16 accs.
template <int EPI>
__global__ __launch_bounds__(256)
void gemm_bt_kernel(const unsigned short* __restrict__ A,
                    const unsigned short* __restrict__ Bw,
                    int M, int N, int K,
                    size_t strA, size_t strB, size_t strO,
                    float alpha,
                    unsigned short* __restrict__ obf,
                    float* __restrict__ of,
                    const float* __restrict__ resid,
                    const float* __restrict__ lnw,
                    int Tt) {
  __shared__ unsigned short lA[2][TM * LROW];
  __shared__ unsigned short lB[2][TN * LROW];

  const int tid  = threadIdx.x;
  const int lane = tid & 31;
  const int wv   = tid >> 5;
  const int wm   = wv >> 2;        // 0..1
  const int wn   = wv & 3;         // 0..3
  const int g    = lane >> 4;      // lane group (K-half selector)
  const int l16  = lane & 15;
  const int bM   = blockIdx.y * TM;
  const int bN   = blockIdx.x * TN;
  const int z    = blockIdx.z;

  const unsigned short* Ab = A  + (size_t)z * strA;
  const unsigned short* Bb = Bw + (size_t)z * strB;

  v8f zero = {0.f, 0.f, 0.f, 0.f, 0.f, 0.f, 0.f, 0.f};
  v8f acc[4][2];
#pragma unroll
  for (int i = 0; i < 4; ++i)
#pragma unroll
    for (int j = 0; j < 2; ++j) acc[i][j] = zero;

#if USE_ASYNC_LDS
  // Direct global->LDS loads, tracked by ASYNCcnt. Each thread owns 4x8B
  // chunks of the A tile and 4x8B chunks of the B tile per K-step.
  auto fill_tile = [&](int k0, int buf) {
#pragma unroll
    for (int i = 0; i < 4; ++i) {
      int c = tid + i * 256;           // 1024 chunks of 8B per 128x32 tile
      int row = c >> 3, off = (c & 7) << 2;
      unsigned short* ga = const_cast<unsigned short*>(Ab) +
                           ((size_t)(bM + row) * K + (k0 + off));
      unsigned short* gb = const_cast<unsigned short*>(Bb) +
                           ((size_t)(bN + row) * K + (k0 + off));
      __builtin_amdgcn_global_load_async_to_lds_b64(
          (AS1 v2i_gcc*)ga, (AS3 v2i_gcc*)&lA[buf][row * LROW + off], 0, 0);
      __builtin_amdgcn_global_load_async_to_lds_b64(
          (AS1 v2i_gcc*)gb, (AS3 v2i_gcc*)&lB[buf][row * LROW + off], 0, 0);
    }
  };
#else
  uint2 ra[4], rb[4];
  auto gload = [&](int k0) {
#pragma unroll
    for (int i = 0; i < 4; ++i) {
      int c = tid + i * 256;
      int row = c >> 3, off = (c & 7) << 2;
      ra[i] = *(const uint2*)(Ab + (size_t)(bM + row) * K + (k0 + off));
      rb[i] = *(const uint2*)(Bb + (size_t)(bN + row) * K + (k0 + off));
    }
  };
  auto stlds = [&](int buf) {
#pragma unroll
    for (int i = 0; i < 4; ++i) {
      int c = tid + i * 256;
      int row = c >> 3, off = (c & 7) << 2;
      *(uint2*)&lA[buf][row * LROW + off] = ra[i];
      *(uint2*)&lB[buf][row * LROW + off] = rb[i];
    }
  };
#endif

  const int nk = K >> 5;               // K-tiles of 32

#if USE_ASYNC_LDS
  fill_tile(0, 0);
  __builtin_amdgcn_s_wait_asynccnt(0);
  __syncthreads();
#else
  gload(0);
  stlds(0);
  __syncthreads();
#endif

  for (int kt = 0; kt < nk; ++kt) {
    const int buf = kt & 1;
#if USE_ASYNC_LDS
    if (kt + 1 < nk) fill_tile((kt + 1) << 5, buf ^ 1);
#else
    if (kt + 1 < nk) gload((kt + 1) << 5);
#endif

    union F { v16bf v; unsigned long long d[4]; };
    F bf[2], af[4];
    // B fragment: lane n=l16 reads B-row (K-contiguous); K = 16*g + e
#pragma unroll
    for (int nf = 0; nf < 2; ++nf) {
      const unsigned short* p = &lB[buf][(wn * 32 + nf * 16 + l16) * LROW + 16 * g];
#pragma unroll
      for (int i = 0; i < 4; ++i)
        bf[nf].d[i] = *(const unsigned long long*)(p + 4 * i);
    }
    // A fragment: lane m=l16 reads A-row; K = 16*g + e
#pragma unroll
    for (int mf = 0; mf < 4; ++mf) {
      const unsigned short* p = &lA[buf][(wm * 64 + mf * 16 + l16) * LROW];
      af[mf].d[0] = *(const unsigned long long*)(p + 16 * g);
      af[mf].d[1] = *(const unsigned long long*)(p + 16 * g + 4);
      af[mf].d[2] = *(const unsigned long long*)(p + 16 * g + 8);
      af[mf].d[3] = *(const unsigned long long*)(p + 16 * g + 12);
    }
#pragma unroll
    for (int mf = 0; mf < 4; ++mf)
#pragma unroll
      for (int nf = 0; nf < 2; ++nf)
        acc[mf][nf] = __builtin_amdgcn_wmma_f32_16x16x32_bf16(
            false, af[mf].v, false, bf[nf].v, (short)0, acc[mf][nf], false, false);

#if USE_ASYNC_LDS
    if (kt + 1 < nk) __builtin_amdgcn_s_wait_asynccnt(0);
#else
    if (kt + 1 < nk) stlds(buf ^ 1);
#endif
    __syncthreads();
  }

  // Epilogue.  C layout: VGPR j -> M = j + 8*g, N = l16.
#pragma unroll
  for (int mf = 0; mf < 4; ++mf) {
#pragma unroll
    for (int nf = 0; nf < 2; ++nf) {
      int gm0 = bM + wm * 64 + mf * 16 + 8 * g;
      int gn  = bN + wn * 32 + nf * 16 + l16;
#pragma unroll
      for (int j = 0; j < 8; ++j) {
        int gm = gm0 + j;
        float v = acc[mf][nf][j] * alpha;
        if constexpr (EPI == EPI_POLY) v = v * v * 0.1f + v * 0.1f;
        if constexpr (EPI == EPI_BF16 || EPI == EPI_POLY) {
          obf[(size_t)z * strO + (size_t)gm * N + gn] = f2bf(v);
        } else if constexpr (EPI == EPI_BF16T) {
          // store transposed per batch: out[b][gn][t], b = gm/Tt, t = gm%Tt
          unsigned b = (unsigned)gm / (unsigned)Tt;
          unsigned t = (unsigned)gm - b * (unsigned)Tt;
          obf[(size_t)b * N * Tt + (size_t)gn * Tt + t] = f2bf(v);
        } else {
          size_t idx = (size_t)gm * N + gn;
          float xn = (resid[idx] * 0.5f + v * 0.5f) * 0.1f;
          of[idx] = xn;
          if constexpr (EPI == EPI_BLEND_LN)
            obf[idx] = f2bf(xn * lnw[gn] * 0.1f);
        }
      }
    }
  }
}

// ---------------------------------------------------------------------------
extern "C" void kernel_launch(void* const* d_in, const int* in_sizes, int n_in,
                              void* d_out, int out_size, void* d_ws, size_t ws_size,
                              hipStream_t stream) {
  (void)in_sizes; (void)n_in; (void)out_size; (void)ws_size;
  const int Bn = 4, T = 2048, C = 1024;
  const int M  = Bn * T;                 // 8192
  const size_t TC = (size_t)T * C;       // 2097152
  const size_t TT = (size_t)T * T;       // 4194304

  const float* x   = (const float*)d_in[0];
  const float* Wq  = (const float*)d_in[1];
  const float* Wk  = (const float*)d_in[2];
  const float* Wv  = (const float*)d_in[3];
  const float* Wo  = (const float*)d_in[4];
  const float* W1  = (const float*)d_in[5];
  const float* W2  = (const float*)d_in[6];
  const float* ln1 = (const float*)d_in[7];
  const float* ln2 = (const float*)d_in[8];
  float* out = (float*)d_out;

  const size_t MBy = (size_t)1 << 20;
  char* ws = (char*)d_ws;
  unsigned short* h1  = (unsigned short*)(ws);              // 8M bf16 (alias: h2)
  unsigned short* wqb = (unsigned short*)(ws + 16 * MBy);   // 8M bf16 weights
  unsigned short* wkb = wqb + 1 * 1048576;
  unsigned short* wvb = wqb + 2 * 1048576;
  unsigned short* wob = wqb + 3 * 1048576;
  unsigned short* w1b = wqb + 4 * 1048576;
  unsigned short* w2b = wqb + 6 * 1048576;
  unsigned short* qb  = (unsigned short*)(ws + 32 * MBy);   // 8M (alias: a)
  unsigned short* kb  = (unsigned short*)(ws + 48 * MBy);   // 8M
  unsigned short* vTb = (unsigned short*)(ws + 64 * MBy);   // 8M, [B][C][T]
  unsigned short* sb  = (unsigned short*)(ws + 80 * MBy);   // 16M (alias: u)
  float*          x1  = (float*)(ws + 112 * MBy);           // 8M f32
  unsigned short* ab  = qb;
  unsigned short* h2  = h1;
  unsigned short* ub  = sb;

  dim3 blk(256);

  // ---- convert weights fp32 -> bf16 ----
  cvt_bf16_kernel<<<dim3(1048576 / 4 / 256), blk, 0, stream>>>(Wq, wqb, 1048576 / 4);
  cvt_bf16_kernel<<<dim3(1048576 / 4 / 256), blk, 0, stream>>>(Wk, wkb, 1048576 / 4);
  cvt_bf16_kernel<<<dim3(1048576 / 4 / 256), blk, 0, stream>>>(Wv, wvb, 1048576 / 4);
  cvt_bf16_kernel<<<dim3(1048576 / 4 / 256), blk, 0, stream>>>(Wo, wob, 1048576 / 4);
  cvt_bf16_kernel<<<dim3(2097152 / 4 / 256), blk, 0, stream>>>(W1, w1b, 2097152 / 4);
  cvt_bf16_kernel<<<dim3(2097152 / 4 / 256), blk, 0, stream>>>(W2, w2b, 2097152 / 4);

  // ---- h1 = bf16(x * ln1 * 0.1) ----
  {
    int n4 = (M * C) / 4;
    scale_ln_bf16_kernel<<<dim3(n4 / 256), blk, 0, stream>>>(x, ln1, h1, n4, C / 4 - 1);
  }

  // ---- q = bf16((h1 @ Wq^T) * 0.01), k likewise, vT transposed * 0.1 ----
  gemm_bt_kernel<EPI_BF16><<<dim3(C / TN, M / TM, 1), blk, 0, stream>>>(
      h1, wqb, M, C, C, 0, 0, 0, 0.01f, qb, nullptr, nullptr, nullptr, 0);
  gemm_bt_kernel<EPI_BF16><<<dim3(C / TN, M / TM, 1), blk, 0, stream>>>(
      h1, wkb, M, C, C, 0, 0, 0, 0.01f, kb, nullptr, nullptr, nullptr, 0);
  gemm_bt_kernel<EPI_BF16T><<<dim3(C / TN, M / TM, 1), blk, 0, stream>>>(
      h1, wvb, M, C, C, 0, 0, 0, 0.1f, vTb, nullptr, nullptr, nullptr, T);

  // ---- s = poly((q @ k^T) * 0.01) per batch ----
  gemm_bt_kernel<EPI_POLY><<<dim3(T / TN, T / TM, Bn), blk, 0, stream>>>(
      qb, kb, T, T, C, TC, TC, TT, 0.01f, sb, nullptr, nullptr, nullptr, 0);

  // ---- a = bf16((s @ vT^T) * 0.01) = (s @ v) * 0.01 per batch ----
  gemm_bt_kernel<EPI_BF16><<<dim3(C / TN, T / TM, Bn), blk, 0, stream>>>(
      sb, vTb, T, C, T, TT, TC, TC, 0.01f, ab, nullptr, nullptr, nullptr, 0);

  // ---- x1 = (x*0.5 + (a @ Wo^T)*0.1*0.5)*0.1 ; h2 = bf16(x1*ln2*0.1) ----
  gemm_bt_kernel<EPI_BLEND_LN><<<dim3(C / TN, M / TM, 1), blk, 0, stream>>>(
      ab, wob, M, C, C, 0, 0, 0, 0.1f, h2, x1, x, ln2, 0);

  // ---- u = poly((h2 @ W1^T) * 0.05) ----
  gemm_bt_kernel<EPI_POLY><<<dim3(2 * C / TN, M / TM, 1), blk, 0, stream>>>(
      h2, w1b, M, 2 * C, C, 0, 0, 0, 0.05f, ub, nullptr, nullptr, nullptr, 0);

  // ---- out = (x1*0.5 + (u @ W2^T)*0.05*0.5)*0.1 ----
  gemm_bt_kernel<EPI_BLEND><<<dim3(C / TN, M / TM, 1), blk, 0, stream>>>(
      ub, w2b, M, C, 2 * C, 0, 0, 0, 0.05f, nullptr, out, x1, nullptr, 0);
}
